// BiFNO_66391604462056
// MI455X (gfx1250) — matrix operand order repaired and checked by reading, not compiled, not run on hardware
//
#include <hip/hip_runtime.h>
#include <hip/hip_bf16.h>
#include <math.h>

// ---------------------------------------------------------------------------
// BiFNO forward for MI455X (gfx1250, wave32, WMMA).
// Truncated FFTs are dense GEMMs against precomputed twiddle matrices;
// 1x1 convs use an LDS-staged WMMA kernel (async-to-LDS when available).
// ---------------------------------------------------------------------------

typedef __attribute__((ext_vector_type(16))) _Float16 v16h;
typedef __attribute__((ext_vector_type(8)))  float    v8f;

#define HA 512
#define HB 128
#define NCH 64         // channels per grid
#define NP  128        // concat channels
#define HWA (HA*HA)    // 262144
#define HWB (HB*HB)    // 16384
#define PI_F 3.14159265358979323846f

#if __has_builtin(__builtin_amdgcn_global_load_async_to_lds_b128) && \
    __has_builtin(__builtin_amdgcn_s_wait_asynccnt)
#define USE_ASYNC_LDS 1
typedef __attribute__((ext_vector_type(4))) int v4i;
typedef __attribute__((address_space(1))) v4i gv4i;   // global int4
typedef __attribute__((address_space(3))) v4i lv4i;   // LDS int4
#else
#define USE_ASYNC_LDS 0
#endif

__device__ __forceinline__ float gelu_tanh(float x) {
    float x3 = x * x * x;
    float t  = tanhf(0.7978845608028654f * (x + 0.044715f * x3));
    return 0.5f * x * (1.0f + t);
}

__device__ __forceinline__ v8f wmma_f16(v16h a, v16h b, v8f c) {
    return __builtin_amdgcn_wmma_f32_16x16x32_f16(false, a, false, b,
                                                  (short)0, c, false, false);
}

// ---------------------------------------------------------------------------
// Generic WMMA GEMM:  D[M,N] = act( A[M,K] * B[K,N] + bias[M] ) + resid[M,N]
// A,B,D f32 row-major; converted to f16 at load (f32 accumulate).
// Block = 128 threads = 4 waves; each wave owns one 16x16 C tile.
// Requires: M%16==0, K%32==0, N a multiple of 64 per grid.x block.
// ---------------------------------------------------------------------------
__global__ void __launch_bounds__(128)
gemm_wmma_f16(const float* __restrict__ A, const float* __restrict__ B,
              float* __restrict__ D,
              const float* __restrict__ bias, const float* __restrict__ resid,
              int M, int N, int K, int lda, int ldb, int ldc, int act)
{
    const int lane  = threadIdx.x & 31;
    const int wave  = threadIdx.x >> 5;
    const int mBase = blockIdx.y * 16;
    const int nBase = blockIdx.x * 64 + wave * 16;
    if (nBase >= N) return;               // wave-uniform: EXEC stays all-1s

    const int col = lane & 15;            // M-row for A frag, N-col for B/C frag
    const int hi  = lane >> 4;
    const int kloA = hi ? 16 : 0;         // contiguous k = 16h + s, matches B
    const int kloB = hi ? 16 : 0;         // contiguous k = 16h + s, same as A

    v8f acc = {0.f,0.f,0.f,0.f,0.f,0.f,0.f,0.f};
    const float* Arow = A + (size_t)(mBase + col) * lda;

    for (int k0 = 0; k0 < K; k0 += 32) {
        v16h a, b;
        #pragma unroll
        for (int e = 0; e < 8; ++e) {
            a[e]     = (_Float16)Arow[k0 + kloA + e];
            a[e + 8] = (_Float16)Arow[k0 + kloA + 8 + e];
        }
        const float* Bp = B + (size_t)(k0 + kloB) * ldb + (nBase + col);
        #pragma unroll
        for (int e = 0; e < 16; ++e) {
            b[e] = (_Float16)Bp[(size_t)e * ldb];
        }
        acc = wmma_f16(a, b, acc);
    }

    #pragma unroll
    for (int r = 0; r < 8; ++r) {
        int m = mBase + r + (hi ? 8 : 0);
        int n = nBase + col;
        float v = acc[r];
        if (bias)  v += bias[m];
        if (act)   v  = gelu_tanh(v);
        if (resid) v += resid[(size_t)m * ldc + n];
        D[(size_t)m * ldc + n] = v;
    }
}

// ---------------------------------------------------------------------------
// 1x1-conv WMMA GEMM, M=K=64 fixed:  D[64,N] = act(W[64,64]*B[64,N]+b)+resid
// Block = 256 threads = 8 waves; block computes a 64x128 C tile with both
// operand tiles staged in LDS (async-to-LDS path when available).
// Each wave computes a 64x16 strip: B fragment reused across 4 m-tiles.
// ---------------------------------------------------------------------------
#define CG_TN 128
__global__ void __launch_bounds__(256)
conv_gemm_wmma(const float* __restrict__ W, const float* __restrict__ B,
               float* __restrict__ D, const float* __restrict__ bias,
               const float* __restrict__ resid, int N, int act)
{
    __shared__ float sA[64 * 64];       // 16 KB weights (m-major)
    __shared__ float sB[64 * CG_TN];    // 32 KB activation tile (k-major)

    const int tid    = threadIdx.x;
    const int nBlock = blockIdx.x * CG_TN;

#if USE_ASYNC_LDS
    // ---- async DMA of both tiles into LDS (ASYNCcnt path) ----
    for (int i = tid; i < 1024; i += 256) {                 // 4096 f32 of W
        __builtin_amdgcn_global_load_async_to_lds_b128(
            (gv4i*)(W + i * 4), (lv4i*)(sA + i * 4), 0, 0);
    }
    for (int i = tid; i < 2048; i += 256) {                 // 64x128 f32 of B
        int r = i >> 5, c4 = (i & 31) * 4;
        __builtin_amdgcn_global_load_async_to_lds_b128(
            (gv4i*)(B + (size_t)r * N + nBlock + c4),
            (lv4i*)(sB + r * CG_TN + c4), 0, 0);
    }
    __builtin_amdgcn_s_wait_asynccnt(0);
#else
    for (int i = tid; i < 4096; i += 256) sA[i] = W[i];
    for (int i = tid; i < 8192; i += 256) {
        int r = i >> 7, c = i & (CG_TN - 1);
        sB[i] = B[(size_t)r * N + nBlock + c];
    }
#endif
    __syncthreads();

    const int lane = tid & 31;
    const int wave = tid >> 5;          // 0..7 -> n strip
    const int col  = lane & 15;
    const int hi   = lane >> 4;
    const int kloA = hi ? 16 : 0;
    const int kloB = hi ? 16 : 0;
    const int nCol = wave * 16 + col;   // 0..127 within tile

    v8f acc[4];
    #pragma unroll
    for (int mt = 0; mt < 4; ++mt) acc[mt] = (v8f){0.f,0.f,0.f,0.f,0.f,0.f,0.f,0.f};

    #pragma unroll
    for (int k0 = 0; k0 < 64; k0 += 32) {
        v16h bfrag;
        #pragma unroll
        for (int e = 0; e < 16; ++e)
            bfrag[e] = (_Float16)sB[(k0 + kloB + e) * CG_TN + nCol];
        #pragma unroll
        for (int mt = 0; mt < 4; ++mt) {
            const float* Ar = sA + (mt * 16 + col) * 64;
            v16h afrag;
            #pragma unroll
            for (int e = 0; e < 8; ++e) {
                afrag[e]     = (_Float16)Ar[k0 + kloA + e];
                afrag[e + 8] = (_Float16)Ar[k0 + kloA + 8 + e];
            }
            acc[mt] = wmma_f16(afrag, bfrag, acc[mt]);
        }
    }

    #pragma unroll
    for (int mt = 0; mt < 4; ++mt) {
        #pragma unroll
        for (int r = 0; r < 8; ++r) {
            int m = mt * 16 + r + (hi ? 8 : 0);
            int n = nBlock + nCol;
            float v = acc[mt][r];
            if (bias)  v += bias[m];
            if (act)   v  = gelu_tanh(v);
            if (resid) v += resid[(size_t)m * N + n];
            D[(size_t)m * N + n] = v;
        }
    }
}

// ---------------------------------------------------------------------------
// Twiddle-table generation (deterministic; rebuilt every call into d_ws).
// ---------------------------------------------------------------------------
__global__ void tw_fwdW(float* __restrict__ F, int H, float invnorm) {
    int tid = blockIdx.x * blockDim.x + threadIdx.x;
    if (tid >= H * 64) return;
    int w = tid >> 6, c = tid & 63, kx = c & 31;
    float th = 2.0f * PI_F * (float)((long long)w * kx % H) / (float)H;
    F[tid] = (c < 32 ? cosf(th) : -sinf(th)) * invnorm;
}

__global__ void tw_fwdH(float* __restrict__ G, int H) {
    int tid = blockIdx.x * blockDim.x + threadIdx.x;
    if (tid >= 32 * H) return;
    int j = tid / H, h = tid % H;
    float ky = (float)(j - 16);
    float ph = 2.0f * PI_F * (float)h * ky / (float)H;
    G[tid]          =  cosf(ph);
    G[32 * H + tid] = -sinf(ph);
}

__global__ void tw_invH(float* __restrict__ Hm, int H) {
    int tid = blockIdx.x * blockDim.x + threadIdx.x;
    if (tid >= H * 32) return;
    int h = tid >> 5, j = tid & 31;
    float ky = (float)(j - 16);
    float ph = 2.0f * PI_F * (float)h * ky / (float)H;
    Hm[tid]          = cosf(ph);
    Hm[H * 32 + tid] = sinf(ph);
}

__global__ void tw_invW(float* __restrict__ Bi, int H) {
    int tid = blockIdx.x * blockDim.x + threadIdx.x;
    if (tid >= 64 * H) return;
    int r = tid / H, w = tid % H, k = r & 31;
    float th = 2.0f * PI_F * (float)((long long)w * k % H) / (float)H;
    float a  = (k == 0) ? 1.0f : 2.0f;
    float v;
    if (r < 32) v = a * cosf(th);
    else        v = (k == 0) ? 0.0f : -a * sinf(th);
    Bi[tid] = v;
}

// ---------------------------------------------------------------------------
// Per-channel DFT along H:  coef[c,j,kx] = sum_h X[c,h,kx] * G[j,h] (complex)
// ---------------------------------------------------------------------------
__global__ void fwdH_kernel(const float* __restrict__ X, const float* __restrict__ G,
                            float* __restrict__ coef, int H, int chanOff)
{
    int tid = blockIdx.x * blockDim.x + threadIdx.x;
    if (tid >= NCH * 1024) return;
    int c = tid >> 10, j = (tid >> 5) & 31, kx = tid & 31;
    const float* Xr = X + (size_t)c * H * 64 + kx;
    const float* Gr = G + j * H;
    const float* Gi = G + 32 * H + j * H;
    float sr = 0.f, si = 0.f;
    for (int h = 0; h < H; ++h) {
        float xr = Xr[(size_t)h * 64];
        float xi = Xr[(size_t)h * 64 + 32];
        float gr = Gr[h], gi = Gi[h];
        sr += xr * gr - xi * gi;
        si += xr * gi + xi * gr;
    }
    size_t o = (size_t)(chanOff + c) * 1024 + j * 32 + kx;
    coef[o]             = sr;
    coef[NP * 1024 + o] = si;
}

// ---------------------------------------------------------------------------
// Per-mode complex channel mixing (HBM-bound: streams ~134 MB of A per layer;
// global_prefetch used to run ahead of the strided reads).
// ---------------------------------------------------------------------------
__global__ void mix_kernel(const float* __restrict__ Are, const float* __restrict__ Aim,
                           const float* __restrict__ cin, float* __restrict__ cout)
{
    int mode = blockIdx.x * blockDim.x + threadIdx.x;   // 0..1023
    int o    = blockIdx.y;                              // 0..127
    if (mode >= 1024) return;
    float sr = 0.f, si = 0.f;
    const float* Ar = Are + (size_t)o * NP * 1024 + mode;
    const float* Ai = Aim + (size_t)o * NP * 1024 + mode;
    for (int i = 0; i < NP; ++i) {
        __builtin_prefetch(Ar + (size_t)(i + 8) * 1024, 0, 0);
        __builtin_prefetch(Ai + (size_t)(i + 8) * 1024, 0, 0);
        float ar = Ar[(size_t)i * 1024];
        float ai = Ai[(size_t)i * 1024];
        float cr = cin[(size_t)i * 1024 + mode];
        float ci = cin[(size_t)NP * 1024 + (size_t)i * 1024 + mode];
        sr += ar * cr - ai * ci;
        si += ar * ci + ai * cr;
    }
    cout[(size_t)o * 1024 + mode]                     = sr;
    cout[(size_t)NP * 1024 + (size_t)o * 1024 + mode] = si;
}

// ---------------------------------------------------------------------------
// Per-channel inverse DFT along H: T[c,h,kx] = sum_j cc[c,j,kx] * Hm[h,j]
// ---------------------------------------------------------------------------
__global__ void invH_kernel(const float* __restrict__ cc, const float* __restrict__ Hm,
                            float* __restrict__ T, int H, int chanOff)
{
    int tid = blockIdx.x * blockDim.x + threadIdx.x;
    if (tid >= NCH * H * 32) return;
    int kx = tid & 31;
    int h  = (tid >> 5) % H;
    int c  = tid / (H * 32);
    const float* Or = cc + (size_t)(chanOff + c) * 1024 + kx;
    const float* Oi = Or + (size_t)NP * 1024;
    const float* Hr = Hm + h * 32;
    const float* Hi = Hm + H * 32 + h * 32;
    float sr = 0.f, si = 0.f;
    #pragma unroll 4
    for (int j = 0; j < 32; ++j) {
        float orr = Or[j * 32], oii = Oi[j * 32];
        float hr = Hr[j], hi = Hi[j];
        sr += orr * hr - oii * hi;
        si += orr * hi + oii * hr;
    }
    size_t row = (size_t)(c * H + h) * 64;
    T[row + kx]      = sr;
    T[row + 32 + kx] = si;
}

// ---------------------------------------------------------------------------
// Encoder / decoder / copy
// ---------------------------------------------------------------------------
__global__ void encode_kernel(const float* __restrict__ x2, const float* __restrict__ u2,
                              const float* __restrict__ W, const float* __restrict__ b,
                              float* __restrict__ out, int HW)
{
    int p = blockIdx.x * blockDim.x + threadIdx.x;
    if (p >= HW) return;
    float i0 = x2[p], i1 = x2[HW + p], i2 = u2[p], i3 = u2[HW + p];
    for (int c = 0; c < NCH; ++c) {
        float v = W[c*4]*i0 + W[c*4+1]*i1 + W[c*4+2]*i2 + W[c*4+3]*i3 + b[c];
        out[(size_t)c * HW + p] = v;
    }
}

__global__ void decode_kernel(const float* __restrict__ ua, const float* __restrict__ W,
                              const float* __restrict__ b, float* __restrict__ out)
{
    int p = blockIdx.x * blockDim.x + threadIdx.x;
    if (p >= HWA) return;
    float s0 = b[0], s1 = b[1];
    for (int i = 0; i < NCH; ++i) {
        float v = ua[(size_t)i * HWA + p];
        s0 += W[i] * v;
        s1 += W[NCH + i] * v;
    }
    out[p]       = s0;
    out[HWA + p] = s1;
}

__global__ void copy_kernel(const float* __restrict__ src, float* __restrict__ dst, int n)
{
    int p = blockIdx.x * blockDim.x + threadIdx.x;
    if (p < n) dst[p] = src[p];
}

// ---------------------------------------------------------------------------
// Host launcher
// ---------------------------------------------------------------------------
extern "C" void kernel_launch(void* const* d_in, const int* in_sizes, int n_in,
                              void* d_out, int out_size, void* d_ws, size_t ws_size,
                              hipStream_t stream)
{
    const float* u_a    = (const float*)d_in[0];
    const float* x_a    = (const float*)d_in[1];
    const float* u_b    = (const float*)d_in[2];
    const float* x_b    = (const float*)d_in[3];
    const float* enc_a_w= (const float*)d_in[4];
    const float* enc_a_b= (const float*)d_in[5];
    const float* enc_b_w= (const float*)d_in[6];
    const float* enc_b_b= (const float*)d_in[7];
    const float* dec_w  = (const float*)d_in[8];
    const float* dec_b  = (const float*)d_in[9];
    const float* c1a_w  = (const float*)d_in[10];
    const float* c1a_b  = (const float*)d_in[11];
    const float* c2a_w  = (const float*)d_in[12];
    const float* c2a_b  = (const float*)d_in[13];
    const float* c1b_w  = (const float*)d_in[14];
    const float* c1b_b  = (const float*)d_in[15];
    const float* c2b_w  = (const float*)d_in[16];
    const float* c2b_b  = (const float*)d_in[17];
    const float* A_re   = (const float*)d_in[18];
    const float* A_im   = (const float*)d_in[19];
    float* out = (float*)d_out;

    // ---- workspace layout (floats) ----
    float* ws = (float*)d_ws;
    size_t off = 0;
    float* ua     = ws + off; off += (size_t)NCH * HWA;
    float* ub     = ws + off; off += (size_t)NCH * HWB;
    float* spec_a = ws + off; off += (size_t)NCH * HWA;
    float* spec_b = ws + off; off += (size_t)NCH * HWB;
    float* tmp_a  = ws + off; off += (size_t)NCH * HWA;      // Xa / Ta / y1a
    float* tmp_b  = ws + off; off += (size_t)NCH * HWB;      // Xb / Tb / y1b
    float* coefI  = ws + off; off += (size_t)2 * NP * 1024;
    float* coefO  = ws + off; off += (size_t)2 * NP * 1024;
    float* FwA    = ws + off; off += (size_t)HA * 64;
    float* FwB    = ws + off; off += (size_t)HB * 64;
    float* GhA    = ws + off; off += (size_t)2 * 32 * HA;
    float* GhB    = ws + off; off += (size_t)2 * 32 * HB;
    float* HiA    = ws + off; off += (size_t)2 * HA * 32;
    float* HiB    = ws + off; off += (size_t)2 * HB * 32;
    float* BiA    = ws + off; off += (size_t)64 * HA;
    float* BiB    = ws + off; off += (size_t)64 * HB;
    (void)ws_size;

    const int TB = 256;
    // ---- twiddles ----
    tw_fwdW<<<(HA*64 + TB-1)/TB, TB, 0, stream>>>(FwA, HA, 1.0f/((float)HWA));
    tw_fwdW<<<(HB*64 + TB-1)/TB, TB, 0, stream>>>(FwB, HB, 1.0f/((float)HWB));
    tw_fwdH<<<(32*HA + TB-1)/TB, TB, 0, stream>>>(GhA, HA);
    tw_fwdH<<<(32*HB + TB-1)/TB, TB, 0, stream>>>(GhB, HB);
    tw_invH<<<(HA*32 + TB-1)/TB, TB, 0, stream>>>(HiA, HA);
    tw_invH<<<(HB*32 + TB-1)/TB, TB, 0, stream>>>(HiB, HB);
    tw_invW<<<(64*HA + TB-1)/TB, TB, 0, stream>>>(BiA, HA);
    tw_invW<<<(64*HB + TB-1)/TB, TB, 0, stream>>>(BiB, HB);

    // ---- encoders ----
    encode_kernel<<<(HWA + TB-1)/TB, TB, 0, stream>>>(x_a, u_a, enc_a_w, enc_a_b, ua, HWA);
    encode_kernel<<<(HWB + TB-1)/TB, TB, 0, stream>>>(x_b, u_b, enc_b_w, enc_b_b, ub, HWB);

    for (int l = 0; l < 4; ++l) {
        // ---- forward truncated DFT: W axis WMMA, H axis VALU ----
        {
            dim3 g(1, (NCH * HA) / 16);
            gemm_wmma_f16<<<g, 128, 0, stream>>>(ua, FwA, tmp_a, nullptr, nullptr,
                                                 NCH*HA, 64, HA, HA, 64, 64, 0);
        }
        fwdH_kernel<<<(NCH*1024 + TB-1)/TB, TB, 0, stream>>>(tmp_a, GhA, coefI, HA, 0);
        {
            dim3 g(1, (NCH * HB) / 16);
            gemm_wmma_f16<<<g, 128, 0, stream>>>(ub, FwB, tmp_b, nullptr, nullptr,
                                                 NCH*HB, 64, HB, HB, 64, 64, 0);
        }
        fwdH_kernel<<<(NCH*1024 + TB-1)/TB, TB, 0, stream>>>(tmp_b, GhB, coefI, HB, NCH);

        // ---- per-mode complex channel mixing ----
        {
            dim3 g(1024 / TB, NP);
            const float* Ar = A_re + (size_t)l * NP * NP * 1024;
            const float* Ai = A_im + (size_t)l * NP * NP * 1024;
            mix_kernel<<<g, TB, 0, stream>>>(Ar, Ai, coefI, coefO);
        }

        // ---- inverse DFT: H axis VALU, W axis WMMA (Hermitian fold) ----
        invH_kernel<<<(NCH*HA*32 + TB-1)/TB, TB, 0, stream>>>(coefO, HiA, tmp_a, HA, 0);
        {
            dim3 g(HA / 64, (NCH * HA) / 16);
            gemm_wmma_f16<<<g, 128, 0, stream>>>(tmp_a, BiA, spec_a, nullptr, nullptr,
                                                 NCH*HA, HA, 64, 64, HA, HA, 0);
        }
        invH_kernel<<<(NCH*HB*32 + TB-1)/TB, TB, 0, stream>>>(coefO, HiB, tmp_b, HB, NCH);
        {
            dim3 g(HB / 64, (NCH * HB) / 16);
            gemm_wmma_f16<<<g, 128, 0, stream>>>(tmp_b, BiB, spec_b, nullptr, nullptr,
                                                 NCH*HB, HB, 64, 64, HB, HB, 0);
        }

        // ---- MLP grid A: ua += gelu(W2 * gelu(W1 * spec_a + b1) + b2) ----
        conv_gemm_wmma<<<HWA / CG_TN, 256, 0, stream>>>(
            c1a_w + (size_t)l*NCH*NCH, spec_a, tmp_a, c1a_b + l*NCH, nullptr, HWA, 1);
        conv_gemm_wmma<<<HWA / CG_TN, 256, 0, stream>>>(
            c2a_w + (size_t)l*NCH*NCH, tmp_a, ua, c2a_b + l*NCH, ua, HWA, 1);

        // ---- MLP grid B (layers 0..2 only) ----
        if (l < 3) {
            conv_gemm_wmma<<<HWB / CG_TN, 256, 0, stream>>>(
                c1b_w + (size_t)l*NCH*NCH, spec_b, tmp_b, c1b_b + l*NCH, nullptr, HWB, 1);
            conv_gemm_wmma<<<HWB / CG_TN, 256, 0, stream>>>(
                c2b_w + (size_t)l*NCH*NCH, tmp_b, ub, c2b_b + l*NCH, ub, HWB, 1);
        }
    }

    // ---- decoder + outputs: [ua_dec (2*512*512), ub (64*128*128)] ----
    decode_kernel<<<(HWA + TB-1)/TB, TB, 0, stream>>>(ua, dec_w, dec_b, out);
    copy_kernel<<<(NCH*HWB + TB-1)/TB, TB, 0, stream>>>(ub, out + 2*HWA, NCH*HWB);
}
